// LlamaMLP_86139864088899
// MI455X (gfx1250) — matrix operand order repaired and checked by reading, not compiled, not run on hardware
//
#include <hip/hip_runtime.h>
#include <hip/hip_bf16.h>
#include <cstdint>

// ---------------------------------------------------------------------------
// Mixed-precision LlamaMLP for MI455X (gfx1250).
// 16 fake-quant GEMMs per linear merged (exactly, by linearity) into ONE f16
// WMMA GEMM per linear with K-concatenated operands. Memory-bound
// (~0.5 GB @ 23.3 TB/s): operands packed to f16, B lives in 192MB L2,
// GEMM tiles staged via GLOBAL_LOAD_ASYNC_TO_LDS_B128 with double buffering.
// ---------------------------------------------------------------------------

typedef __attribute__((ext_vector_type(16))) _Float16     v16h;
typedef __attribute__((ext_vector_type(4)))  _Float16     v4h;
typedef __attribute__((ext_vector_type(8)))  float        v8f;
typedef __attribute__((ext_vector_type(4)))  unsigned int v4u;

#define HS0 1024
#define HS1 2048
#define IS0 2752
#define IS1 5504
#define M_TOK 1024
#define KA_UP (HS0 + HS1)   // 3072
#define KA_DN (IS0 + IS1)   // 8256

// params layout (float offsets inside prm[])
#define P_PA  0    // p_in[i]*p_a[k]      [4]
#define P_PB  4    // p_out[j]*p_w[l]     [4]
#define P_SX  8    // x scales  [i*2+k]   [4]
#define P_SH  12   // h scales  [i*2+k]   [4]
#define P_SWU 16   // W_up   scales [i*4+j*2+l] [8]
#define P_SWG 24   // W_gate scales              [8]
#define P_SWD 32   // W_down scales              [8]

__device__ __forceinline__ float rq(float v, float s) {
  return rintf(v / s) * s;   // round-half-even, matches jnp.round
}

// CDNA5 async memory->LDS copy (16B per lane), tracked by ASYNCcnt.
__device__ __forceinline__ void async_copy16(uint32_t lds_addr, const void* gptr) {
  asm volatile("global_load_async_to_lds_b128 %0, %1, off"
               :: "v"(lds_addr), "v"((uint64_t)(uintptr_t)gptr)
               : "memory");
}
__device__ __forceinline__ void wait_async0() {
  asm volatile("s_wait_asynccnt 0x0" ::: "memory");
}

// ---------------------------------------------------------------------------
__global__ void k_init(unsigned* stat) {
  if (threadIdx.x < 16) stat[threadIdx.x] = 0u;
}

// quadrant abs-max: slot = (row<rowSplit?0:2) + (col<colSplit?0:1)
// float4 per thread; split boundaries are multiples of 4.
__global__ __launch_bounds__(256) void k_absmax(const float* __restrict__ src,
                                                int cols, int rowSplit, int colSplit,
                                                unsigned* __restrict__ stat) {
  const int r = blockIdx.y;
  const int tid = threadIdx.x;
  float m0 = 0.f, m1 = 0.f;
  for (int c = (blockIdx.x * 256 + tid) * 4; c < cols; c += gridDim.x * 1024) {
    float4 v = *(const float4*)(src + (size_t)r * cols + c);
    float mm = fmaxf(fmaxf(fabsf(v.x), fabsf(v.y)), fmaxf(fabsf(v.z), fabsf(v.w)));
    if (c < colSplit) m0 = fmaxf(m0, mm); else m1 = fmaxf(m1, mm);
  }
  __shared__ float s0[256], s1[256];
  s0[tid] = m0; s1[tid] = m1;
  __syncthreads();
  for (int s = 128; s > 0; s >>= 1) {
    if (tid < s) { s0[tid] = fmaxf(s0[tid], s0[tid + s]); s1[tid] = fmaxf(s1[tid], s1[tid + s]); }
    __syncthreads();
  }
  if (tid == 0) {
    const int base = (r < rowSplit) ? 0 : 2;
    atomicMax(&stat[base + 0], __float_as_uint(s0[0]));
    atomicMax(&stat[base + 1], __float_as_uint(s1[0]));
  }
}

__device__ __forceinline__ void softmax2(const float* p, float* w) {
  float m = fmaxf(p[0], p[1]);
  float e0 = expf(p[0] - m), e1 = expf(p[1] - m);
  float s = e0 + e1;
  w[0] = e0 / s; w[1] = e1 / s;
}

__global__ void k_prep1(const float* ph, const float* pi, const float* pa,
                        const float* pw, const unsigned* stat, float* prm) {
  float wh[2], wi[2], wa[2], ww[2];
  softmax2(ph, wh); softmax2(pi, wi); softmax2(pa, wa); softmax2(pw, ww);
  for (int i = 0; i < 2; ++i)
    for (int k = 0; k < 2; ++k) prm[P_PA + i * 2 + k] = wh[i] * wa[k];
  for (int j = 0; j < 2; ++j)
    for (int l = 0; l < 2; ++l) prm[P_PB + j * 2 + l] = wi[j] * ww[l];
  const float qm[2] = {7.f, 127.f};
  float mx0 = __uint_as_float(stat[0]);
  float mxc[2] = {mx0, fmaxf(mx0, __uint_as_float(stat[1]))};
  for (int i = 0; i < 2; ++i)
    for (int k = 0; k < 2; ++k) prm[P_SX + i * 2 + k] = fmaxf(mxc[i] / qm[k], 1e-8f);
  const int sb[3] = {2, 6, 10};
  const int po[3] = {P_SWU, P_SWG, P_SWD};
  for (int w = 0; w < 3; ++w) {
    float q0 = __uint_as_float(stat[sb[w] + 0]);
    float q1 = __uint_as_float(stat[sb[w] + 1]);
    float q2 = __uint_as_float(stat[sb[w] + 2]);
    float q3 = __uint_as_float(stat[sb[w] + 3]);
    float Mij[2][2];
    Mij[0][0] = q0;
    Mij[1][0] = fmaxf(q0, q1);
    Mij[0][1] = fmaxf(q0, q2);
    Mij[1][1] = fmaxf(fmaxf(q0, q1), fmaxf(q2, q3));
    for (int i = 0; i < 2; ++i)
      for (int j = 0; j < 2; ++j)
        for (int l = 0; l < 2; ++l)
          prm[po[w] + i * 4 + j * 2 + l] = fmaxf(Mij[i][j] / qm[l], 1e-8f);
  }
}

__global__ void k_prep2(const unsigned* stat, float* prm) {
  const float qm[2] = {7.f, 127.f};
  float mh0 = __uint_as_float(stat[14]);
  float mhc[2] = {mh0, fmaxf(mh0, __uint_as_float(stat[15]))};
  for (int i = 0; i < 2; ++i)
    for (int k = 0; k < 2; ++k) prm[P_SH + i * 2 + k] = fmaxf(mhc[i] / qm[k], 1e-8f);
}

// A pack (4 cols/thread): dst[m,c] = sum_k p_in[i]p_a[k]*q(v, s[i,k])
__global__ __launch_bounds__(256) void k_pack_A(const float* __restrict__ src,
                                                _Float16* __restrict__ dst,
                                                int srcPitch, int cin0, int KA,
                                                const float* __restrict__ prm, int sOff) {
  const int m  = blockIdx.y;
  const int c4 = (blockIdx.x * 256 + threadIdx.x) * 4;
  if (c4 >= KA) return;
  const int i  = (c4 < cin0) ? 0 : 1;
  const int cs = (i == 0) ? c4 : (c4 - cin0);
  const float4 v = *(const float4*)(src + (size_t)m * srcPitch + cs);
  const float s0 = prm[sOff + i * 2 + 0], s1 = prm[sOff + i * 2 + 1];
  const float p0 = prm[P_PA + i * 2 + 0], p1 = prm[P_PA + i * 2 + 1];
  v4h o;
  o[0] = (_Float16)(p0 * rq(v.x, s0) + p1 * rq(v.x, s1));
  o[1] = (_Float16)(p0 * rq(v.y, s0) + p1 * rq(v.y, s1));
  o[2] = (_Float16)(p0 * rq(v.z, s0) + p1 * rq(v.z, s1));
  o[3] = (_Float16)(p0 * rq(v.w, s0) + p1 * rq(v.w, s1));
  *(v4h*)(dst + (size_t)m * KA + c4) = o;
}

// B pack (4 cols/thread): dst[o,c] = sum_{j>=j0(o)} sum_l p_out[j]p_w[l]*q(w,s[i,j,l])
__global__ __launch_bounds__(256) void k_pack_B(const float* __restrict__ W,
                                                _Float16* __restrict__ dst,
                                                int inPitch, int cin0, int KB, int cout0,
                                                const float* __restrict__ prm, int swOff) {
  const int o  = blockIdx.y;
  const int c4 = (blockIdx.x * 256 + threadIdx.x) * 4;
  if (c4 >= KB) return;
  const int i  = (c4 < cin0) ? 0 : 1;
  const int cs = (i == 0) ? c4 : (c4 - cin0);
  const float4 w = *(const float4*)(W + (size_t)o * inPitch + cs);
  const int j0 = (o < cout0) ? 0 : 1;
  float a0 = 0.f, a1 = 0.f, a2 = 0.f, a3 = 0.f;
  for (int j = j0; j < 2; ++j)
    for (int l = 0; l < 2; ++l) {
      const float p = prm[P_PB + j * 2 + l];
      const float s = prm[swOff + i * 4 + j * 2 + l];
      a0 += p * rq(w.x, s); a1 += p * rq(w.y, s);
      a2 += p * rq(w.z, s); a3 += p * rq(w.w, s);
    }
  v4h ov; ov[0] = (_Float16)a0; ov[1] = (_Float16)a1; ov[2] = (_Float16)a2; ov[3] = (_Float16)a3;
  *(v4h*)(dst + (size_t)o * KB + c4) = ov;
}

// ---------------------------------------------------------------------------
// WMMA GEMM: C[M,N] = A[M,K] * B[N,K]^T, f16 in / f32 acc.
// 256 thr = 8 waves (2x4), BM=BN=128, BK=32; wave owns 4x2 16x16 C tiles.
// Double-buffered LDS tiles filled by async global->LDS copies (ASYNCcnt);
// rows padded to 40 halves -> conflict-free fragment reads.
// ---------------------------------------------------------------------------
union Frag { v16h v; v4u q[2]; };

__global__ __launch_bounds__(256) void k_gemm(const _Float16* __restrict__ A,
                                              const _Float16* __restrict__ B,
                                              float* __restrict__ C,
                                              int N, int K) {
  __shared__ _Float16 sm[2][2][128 * 40];   // [buf][A|B][128 rows x 40 halves]
  const int tid  = threadIdx.x;
  const int lane = tid & 31;
  const int wave = tid >> 5;
  const int wm   = wave >> 2;    // 0..1 : M block of 64
  const int wn   = wave & 3;     // 0..3 : N block of 32
  const int m16  = lane & 15;
  const int half = lane >> 4;
  const int bm0  = blockIdx.y * 128;
  const int bn0  = blockIdx.x * 128;

  // async-copy chunk assignment: 512 16B chunks per tile, 2 per thread
  const int r0 = tid >> 2;             // rows 0..63
  const int r1 = r0 + 64;              // rows 64..127
  const int o0 = (tid & 3) * 8;        // half offset within 32-half row
  const uint32_t lb = (uint32_t)(uintptr_t)&sm[0][0][0];
  const uint32_t lA0 = lb + r0 * 80 + o0 * 2;
  const uint32_t lA1 = lb + r1 * 80 + o0 * 2;
  const _Float16* gA0 = A + (size_t)(bm0 + r0) * K + o0;
  const _Float16* gA1 = A + (size_t)(bm0 + r1) * K + o0;
  const _Float16* gB0 = B + (size_t)(bn0 + r0) * K + o0;
  const _Float16* gB1 = B + (size_t)(bn0 + r1) * K + o0;

  v8f acc[4][2] = {};

  auto loadTile = [&](int kt, int buf) {
    const uint32_t bo = (uint32_t)buf * 20480u;   // bytes per buffer
    async_copy16(lA0 + bo,          gA0 + kt);
    async_copy16(lA1 + bo,          gA1 + kt);
    async_copy16(lA0 + bo + 10240u, gB0 + kt);
    async_copy16(lA1 + bo + 10240u, gB1 + kt);
  };

  loadTile(0, 0);
  wait_async0();
  __syncthreads();

  int buf = 0;
  for (int kt = 0; kt < K; kt += 32) {
    if (kt + 32 < K) loadTile(kt + 32, buf ^ 1);   // prefetch next tile (async)

    const _Float16* tA = sm[buf][0];
    const _Float16* tB = sm[buf][1];
    Frag a[4], b[2];
#pragma unroll
    for (int mi = 0; mi < 4; ++mi) {      // ISA 16-bit A 16x32 layout
      const int r = wm * 64 + mi * 16 + m16;
      a[mi].q[0] = *(const v4u*)(&tA[r * 40 + half * 8]);
      a[mi].q[1] = *(const v4u*)(&tA[r * 40 + 16 + half * 8]);
    }
#pragma unroll
    for (int ni = 0; ni < 2; ++ni) {      // B lane: col n=m16, K=half*8..+7, 16+half*8..+7
      const int r = wn * 32 + ni * 16 + m16;
      b[ni].q[0] = *(const v4u*)(&tB[r * 40 + half * 8]);
      b[ni].q[1] = *(const v4u*)(&tB[r * 40 + 16 + half * 8]);
    }
#pragma unroll
    for (int mi = 0; mi < 4; ++mi)
#pragma unroll
      for (int ni = 0; ni < 2; ++ni)
        acc[mi][ni] = __builtin_amdgcn_wmma_f32_16x16x32_f16(
            false, a[mi].v, false, b[ni].v, (short)0, acc[mi][ni], false, false);

    wait_async0();      // next buffer fully in LDS
    __syncthreads();    // everyone done reading current buffer
    buf ^= 1;
  }

  // C/D layout: lane col n=m16, rows half*8 + r
#pragma unroll
  for (int mi = 0; mi < 4; ++mi)
#pragma unroll
    for (int ni = 0; ni < 2; ++ni) {
      const int gm = bm0 + wm * 64 + mi * 16 + half * 8;
      const int gn = bn0 + wn * 32 + ni * 16 + m16;
#pragma unroll
      for (int r = 0; r < 8; ++r)
        C[(size_t)(gm + r) * N + gn] = acc[mi][ni][r];
    }
}

__global__ __launch_bounds__(256) void k_silu_mul(float* __restrict__ G,
                                                  const float* __restrict__ U, int n4) {
  const int idx = blockIdx.x * 256 + threadIdx.x;
  if (idx < n4) {
    float4 g = ((const float4*)G)[idx];
    float4 u = ((const float4*)U)[idx];
    g.x = (g.x / (1.f + expf(-g.x))) * u.x;
    g.y = (g.y / (1.f + expf(-g.y))) * u.y;
    g.z = (g.z / (1.f + expf(-g.z))) * u.z;
    g.w = (g.w / (1.f + expf(-g.w))) * u.w;
    ((float4*)G)[idx] = g;
  }
}

// ---------------------------------------------------------------------------
extern "C" void kernel_launch(void* const* d_in, const int* in_sizes, int n_in,
                              void* d_out, int out_size, void* d_ws, size_t ws_size,
                              hipStream_t stream) {
  (void)in_sizes; (void)n_in; (void)out_size; (void)ws_size;
  const float* x      = (const float*)d_in[0];
  const float* ph     = (const float*)d_in[1];
  const float* pi     = (const float*)d_in[2];
  const float* pa     = (const float*)d_in[3];
  const float* pw     = (const float*)d_in[4];
  const float* W_gate = (const float*)d_in[5];
  const float* W_up   = (const float*)d_in[6];
  const float* W_down = (const float*)d_in[7];
  float* out = (float*)d_out;

  char* ws = (char*)d_ws;
  unsigned* stat = (unsigned*)ws;                 // 16 quadrant maxes (uint bits)
  float*    prm  = (float*)(ws + 256);            // derived params
  size_t off = 1024;
  _Float16* A1 = (_Float16*)(ws + off); off += (size_t)M_TOK * KA_UP * 2; // 1024x3072
  _Float16* A2 = (_Float16*)(ws + off); off += (size_t)M_TOK * KA_DN * 2; // 1024x8256
  _Float16* Bb = (_Float16*)(ws + off); off += (size_t)IS1 * KA_UP * 2;   // shared B buf
  float*    U  = (float*)(ws + off);    off += (size_t)M_TOK * IS1 * 4;   // 1024x5504
  float*    G  = (float*)(ws + off);                                       // 1024x5504 (->h)

  k_init<<<1, 64, 0, stream>>>(stat);

  k_absmax<<<dim3(2, 1024), 256, 0, stream>>>(x,      HS1, 1024, HS0, stat + 0);
  k_absmax<<<dim3(2, 5504), 256, 0, stream>>>(W_up,   HS1, IS0,  HS0, stat + 2);
  k_absmax<<<dim3(2, 5504), 256, 0, stream>>>(W_gate, HS1, IS0,  HS0, stat + 6);
  k_absmax<<<dim3(6, 2048), 256, 0, stream>>>(W_down, IS1, HS0,  IS0, stat + 10);

  k_prep1<<<1, 1, 0, stream>>>(ph, pi, pa, pw, stat, prm);

  // up projection
  k_pack_A<<<dim3(3, M_TOK), 256, 0, stream>>>(x, A1, HS1, HS0, KA_UP, prm, P_SX);
  k_pack_B<<<dim3(3, IS1), 256, 0, stream>>>(W_up, Bb, HS1, HS0, KA_UP, IS0, prm, P_SWU);
  k_gemm<<<dim3(IS1 / 128, M_TOK / 128), 256, 0, stream>>>(A1, Bb, U, IS1, KA_UP);

  // gate projection (reuse B buffer)
  k_pack_B<<<dim3(3, IS1), 256, 0, stream>>>(W_gate, Bb, HS1, HS0, KA_UP, IS0, prm, P_SWG);
  k_gemm<<<dim3(IS1 / 128, M_TOK / 128), 256, 0, stream>>>(A1, Bb, G, IS1, KA_UP);

  // h = silu(gate) * up   (in place over G)
  k_silu_mul<<<(M_TOK * IS1 / 4 + 255) / 256, 256, 0, stream>>>(G, U, M_TOK * IS1 / 4);

  // down projection
  k_absmax<<<dim3(6, 1024), 256, 0, stream>>>(G, IS1, 1024, IS0, stat + 14);
  k_prep2<<<1, 1, 0, stream>>>(stat, prm);
  k_pack_A<<<dim3(9, M_TOK), 256, 0, stream>>>(G, A2, IS1, IS0, KA_DN, prm, P_SH);
  k_pack_B<<<dim3(9, HS1), 256, 0, stream>>>(W_down, Bb, IS1, IS0, KA_DN, HS0, prm, P_SWD);
  k_gemm<<<dim3(HS1 / 128, M_TOK / 128), 256, 0, stream>>>(A2, Bb, out, HS1, KA_DN);
}
